// STSN_17179869184259
// MI455X (gfx1250) — matrix operand order repaired and checked by reading, not compiled, not run on hardware
//
#include <hip/hip_runtime.h>
#include <hip/hip_bf16.h>

#define HW   9216      // 96*96
#define WDIM 96

typedef __attribute__((ext_vector_type(16))) __bf16 v16bf;
typedef __attribute__((ext_vector_type(8)))  __bf16 v8bf;
typedef __attribute__((ext_vector_type(8)))  float  v8f;

__device__ __forceinline__ unsigned short f32_to_bf16(float f) {
    unsigned int u = __float_as_uint(f);
    unsigned int r = (u + 0x7FFFu + ((u >> 16) & 1u)) >> 16;   // RNE
    return (unsigned short)r;
}

// ---------------------------------------------------------------------------
// Re-pack dcn weights: w[o][c][k] (fp32, k=ky*3+kx) -> A[o][k*C + c] (bf16)
// so the GEMM K-index (k*C+c) matches the im2col column layout.
// ---------------------------------------------------------------------------
__global__ void stsn_pack_w(const float* __restrict__ w, unsigned short* __restrict__ A,
                            int O, int C) {
    int t = blockIdx.x * blockDim.x + threadIdx.x;
    int total = O * C * 9;
    if (t >= total) return;
    int o = t / (C * 9);
    int r = t - o * C * 9;
    int c = r / 9;
    int k = r - c * 9;
    A[(size_t)o * (9 * C) + (size_t)k * C + c] = f32_to_bf16(w[t]);
}

// ---------------------------------------------------------------------------
// Plain 3x3 conv, pad 1: off[oc][p] = sum_c sum_k f[c][p+k] * w[oc][c][k] + b[oc]
// oc = 18 offset channels. fp32 direct (only ~1.5 GFLOP).
// ---------------------------------------------------------------------------
__global__ void stsn_offconv(const float* __restrict__ f, const float* __restrict__ w,
                             const float* __restrict__ b, float* __restrict__ off, int Cin) {
    int t = blockIdx.x * blockDim.x + threadIdx.x;
    if (t >= 18 * HW) return;
    int oc = t / HW;
    int p  = t - oc * HW;
    int y = p / WDIM, x = p - (p / WDIM) * WDIM;
    float s = b[oc];
    const float* wo = w + (size_t)oc * Cin * 9;
    for (int c = 0; c < Cin; ++c) {
        const float* fc = f + (size_t)c * HW;
        const float* wc = wo + c * 9;
#pragma unroll
        for (int ky = 0; ky < 3; ++ky) {
            int yy = y + ky - 1;
            if (yy < 0 || yy >= WDIM) continue;
#pragma unroll
            for (int kx = 0; kx < 3; ++kx) {
                int xx = x + kx - 1;
                if (xx < 0 || xx >= WDIM) continue;
                s = fmaf(fc[yy * WDIM + xx], wc[ky * 3 + kx], s);
            }
        }
    }
    off[t] = s;
}

// ---------------------------------------------------------------------------
// Deformable bilinear im2col -> colT[p][k*C + c] in bf16.
// grid: (HW/64, 9), block: 256 = 64 pixels x 4 channel-groups.
// Gathers stay in L2 (x tensor is <=18 MB, L2 = 192 MB).
// ---------------------------------------------------------------------------
__global__ void stsn_im2col(const float* __restrict__ x, const float* __restrict__ off,
                            unsigned short* __restrict__ colT, int C, int Kdim) {
    int k  = blockIdx.y;                       // 0..8
    int p  = blockIdx.x * 64 + (threadIdx.x & 63);
    int cg = threadIdx.x >> 6;                 // 0..3
    int y = p / WDIM, xx = p - (p / WDIM) * WDIM;
    float dy = off[(size_t)(2 * k)     * HW + p];
    float dx = off[(size_t)(2 * k + 1) * HW + p];
    int ky = k / 3, kx = k - (k / 3) * 3;
    float py = (float)(y - 1 + ky) + dy;
    float px = (float)(xx - 1 + kx) + dx;
    float y0f = floorf(py), x0f = floorf(px);
    float ay = py - y0f, ax = px - x0f;
    int y0 = (int)y0f, x0 = (int)x0f, y1 = y0 + 1, x1 = x0 + 1;
    float vy0 = (y0 >= 0 && y0 < WDIM) ? 1.f : 0.f;
    float vy1 = (y1 >= 0 && y1 < WDIM) ? 1.f : 0.f;
    float vx0 = (x0 >= 0 && x0 < WDIM) ? 1.f : 0.f;
    float vx1 = (x1 >= 0 && x1 < WDIM) ? 1.f : 0.f;
    float w00 = (1.f - ay) * (1.f - ax) * vy0 * vx0;
    float w01 = (1.f - ay) * ax         * vy0 * vx1;
    float w10 = ay         * (1.f - ax) * vy1 * vx0;
    float w11 = ay         * ax         * vy1 * vx1;
    int iy0 = min(max(y0, 0), WDIM - 1), iy1 = min(max(y1, 0), WDIM - 1);
    int ix0 = min(max(x0, 0), WDIM - 1), ix1 = min(max(x1, 0), WDIM - 1);
    int i00 = iy0 * WDIM + ix0, i01 = iy0 * WDIM + ix1;
    int i10 = iy1 * WDIM + ix0, i11 = iy1 * WDIM + ix1;

    int cpg = C >> 2;
    size_t base = (size_t)p * Kdim + (size_t)k * C;
    for (int c = cg * cpg; c < (cg + 1) * cpg; ++c) {
        const float* xc = x + (size_t)c * HW;
        float v = w00 * xc[i00] + w01 * xc[i01] + w10 * xc[i10] + w11 * xc[i11];
        colT[base + c] = f32_to_bf16(v);
    }
}

// ---------------------------------------------------------------------------
// WMMA GEMM: D[O][HW] (fp32) = A[O][K] (bf16) * colT[HW][K]^T (bf16)
// grid: (HW/64, O/128), block 256 = 8 waves; wave w -> rows [w*16, w*16+16),
// each wave computes a 16x64 tile (4 accumulators), K-loop step 32.
// Fragment loads are contiguous b128s; A and B share this contiguous K order:
//   A lane(m=l&15, hi=l>>4): halves [0..15] = K hi*16+j  (contiguous 32B)
//   B lane(n=l&15, hi=l>>4): halves [0..15] = K hi*16+j  (contiguous 32B)
// ---------------------------------------------------------------------------
__global__ void stsn_wmma_gemm(const unsigned short* __restrict__ Aw,
                               const unsigned short* __restrict__ BT,
                               float* __restrict__ D, int K) {
    int lane = threadIdx.x & 31;
    int wv   = threadIdx.x >> 5;       // 0..7
    int hi   = lane >> 4;              // 0 or 1
    int lm   = lane & 15;
    int tileM = blockIdx.y * 128 + wv * 16;
    int tileN = blockIdx.x * 64;

    const __bf16* A = reinterpret_cast<const __bf16*>(Aw);
    const __bf16* B = reinterpret_cast<const __bf16*>(BT);

    v8f acc0 = {}, acc1 = {}, acc2 = {}, acc3 = {};

    const __bf16* arow = A + (size_t)(tileM + lm) * K;
    const __bf16* b0r  = B + (size_t)(tileN + 0  + lm) * K + hi * 16;
    const __bf16* b1r  = B + (size_t)(tileN + 16 + lm) * K + hi * 16;
    const __bf16* b2r  = B + (size_t)(tileN + 32 + lm) * K + hi * 16;
    const __bf16* b3r  = B + (size_t)(tileN + 48 + lm) * K + hi * 16;

    for (int kt = 0; kt < K; kt += 32) {
        v8bf a_lo = *reinterpret_cast<const v8bf*>(arow + kt + hi * 16);
        v8bf a_hi = *reinterpret_cast<const v8bf*>(arow + kt + hi * 16 + 8);
        v16bf a = __builtin_shufflevector(a_lo, a_hi,
                                          0, 1, 2, 3, 4, 5, 6, 7,
                                          8, 9, 10, 11, 12, 13, 14, 15);
        v16bf b0 = *reinterpret_cast<const v16bf*>(b0r + kt);
        v16bf b1 = *reinterpret_cast<const v16bf*>(b1r + kt);
        v16bf b2 = *reinterpret_cast<const v16bf*>(b2r + kt);
        v16bf b3 = *reinterpret_cast<const v16bf*>(b3r + kt);

        acc0 = __builtin_amdgcn_wmma_f32_16x16x32_bf16(false, a, false, b0, (short)0, acc0, false, false);
        acc1 = __builtin_amdgcn_wmma_f32_16x16x32_bf16(false, a, false, b1, (short)0, acc1, false, false);
        acc2 = __builtin_amdgcn_wmma_f32_16x16x32_bf16(false, a, false, b2, (short)0, acc2, false, false);
        acc3 = __builtin_amdgcn_wmma_f32_16x16x32_bf16(false, a, false, b3, (short)0, acc3, false, false);
    }

    // C/D layout: VGPR r -> M = 8*hi + r, N = lane&15
#pragma unroll
    for (int r = 0; r < 8; ++r) {
        size_t row = (size_t)(tileM + 8 * hi + r) * HW;
        D[row + tileN + 0  + lm] = acc0[r];
        D[row + tileN + 16 + lm] = acc1[r];
        D[row + tileN + 32 + lm] = acc2[r];
        D[row + tileN + 48 + lm] = acc3[r];
    }
}

// ---------------------------------------------------------------------------
// GroupNorm (32 groups) + ReLU, in place. One block per group.
// ---------------------------------------------------------------------------
__global__ void stsn_gn_relu(float* __restrict__ data, const float* __restrict__ gamma,
                             const float* __restrict__ beta, int cpg) {
    __shared__ float ssum[256];
    __shared__ float ssq[256];
    int grp = blockIdx.x;
    size_t base = (size_t)grp * cpg * HW;
    int n = cpg * HW;
    float s = 0.f, q = 0.f;
    for (int i = threadIdx.x; i < n; i += 256) {
        float v = data[base + i];
        s += v;
        q += v * v;
    }
    ssum[threadIdx.x] = s;
    ssq[threadIdx.x]  = q;
    __syncthreads();
    for (int st = 128; st > 0; st >>= 1) {
        if ((int)threadIdx.x < st) {
            ssum[threadIdx.x] += ssum[threadIdx.x + st];
            ssq[threadIdx.x]  += ssq[threadIdx.x + st];
        }
        __syncthreads();
    }
    float mean = ssum[0] / (float)n;
    float var  = ssq[0] / (float)n - mean * mean;
    float inv  = rsqrtf(var + 1e-5f);
    for (int i = threadIdx.x; i < n; i += 256) {
        int ch = grp * cpg + i / HW;
        float v = (data[base + i] - mean) * inv * gamma[ch] + beta[ch];
        data[base + i] = v > 0.f ? v : 0.f;
    }
}

// ---------------------------------------------------------------------------
// Final: cos-sim weights + softmax blend. One thread per pixel.
// ---------------------------------------------------------------------------
__global__ void stsn_blend(const float* __restrict__ tt, const float* __restrict__ tk,
                           float* __restrict__ out) {
    int p = blockIdx.x * blockDim.x + threadIdx.x;
    if (p >= HW) return;
    float stt = 0.f, stk = 0.f, sx = 0.f;
    for (int c = 0; c < 256; ++c) {
        float a = tt[(size_t)c * HW + p];
        float b = tk[(size_t)c * HW + p];
        stt += a * a;
        stk += b * b;
        sx  += a * b;
    }
    float ttw = stt / fmaxf(stt, 1e-8f);
    float tkw = sx  / fmaxf(sqrtf(stt) * sqrtf(stk), 1e-8f);
    float m  = fmaxf(ttw, tkw);
    float e0 = __expf(ttw - m), e1 = __expf(tkw - m);
    float rs = 1.f / (e0 + e1);
    float w0 = e0 * rs, w1 = e1 * rs;
    for (int c = 0; c < 256; ++c)
        out[(size_t)c * HW + p] = w0 * tt[(size_t)c * HW + p] + w1 * tk[(size_t)c * HW + p];
}

// ===========================================================================
extern "C" void kernel_launch(void* const* d_in, const int* in_sizes, int n_in,
                              void* d_out, int out_size, void* d_ws, size_t ws_size,
                              hipStream_t stream) {
    const float* datas = (const float*)d_in[0];
    const float* supp  = datas;                      // (256, 96, 96)
    const float* refr  = datas + (size_t)256 * HW;   // (256, 96, 96)

    const float* off_w[4] = { (const float*)d_in[1],  (const float*)d_in[6],
                              (const float*)d_in[11], (const float*)d_in[16] };
    const float* off_b[4] = { (const float*)d_in[2],  (const float*)d_in[7],
                              (const float*)d_in[12], (const float*)d_in[17] };
    const float* dcn_w[4] = { (const float*)d_in[3],  (const float*)d_in[8],
                              (const float*)d_in[13], (const float*)d_in[18] };
    const float* gn_g[4]  = { (const float*)d_in[4],  (const float*)d_in[9],
                              (const float*)d_in[14], (const float*)d_in[19] };
    const float* gn_b[4]  = { (const float*)d_in[5],  (const float*)d_in[10],
                              (const float*)d_in[15], (const float*)d_in[20] };

    // ---- workspace carve ----
    char* wp = (char*)d_ws;
    auto carve = [&](size_t bytes) -> void* {
        void* p = (void*)wp;
        wp += (bytes + 255) & ~(size_t)255;
        return p;
    };
    unsigned short* A123[3];
    for (int i = 0; i < 3; ++i)
        A123[i] = (unsigned short*)carve((size_t)512 * 4608 * 2);
    unsigned short* A4 = (unsigned short*)carve((size_t)256 * 2304 * 2);
    float* f0   = (float*)carve((size_t)512 * HW * 4);
    float* f1   = (float*)carve((size_t)512 * HW * 4);
    float* offb = (float*)carve((size_t)18 * HW * 4);
    unsigned short* colT = (unsigned short*)carve((size_t)HW * 4608 * 2);
    float* ttb  = (float*)carve((size_t)256 * HW * 4);
    float* tkb  = (float*)carve((size_t)256 * HW * 4);

    // ---- pack dcn weights (once; shared by both agg passes) ----
    for (int i = 0; i < 3; ++i) {
        int total = 512 * 512 * 9;
        stsn_pack_w<<<(total + 255) / 256, 256, 0, stream>>>(dcn_w[i], A123[i], 512, 512);
    }
    {
        int total = 256 * 256 * 9;
        stsn_pack_w<<<(total + 255) / 256, 256, 0, stream>>>(dcn_w[3], A4, 256, 256);
    }

    // ---- one agg() pass ----
    auto run_agg = [&](const float* agg_supp, const float* agg_ref, float* out_feat) {
        // f = concat([supp, ref]) -> f0
        hipMemcpyAsync(f0, agg_supp, (size_t)256 * HW * 4, hipMemcpyDeviceToDevice, stream);
        hipMemcpyAsync(f0 + (size_t)256 * HW, agg_ref, (size_t)256 * HW * 4,
                       hipMemcpyDeviceToDevice, stream);
        float* fc = f0;
        float* fn = f1;
        for (int L = 0; L < 3; ++L) {
            stsn_offconv<<<(18 * HW + 255) / 256, 256, 0, stream>>>(fc, off_w[L], off_b[L], offb, 512);
            stsn_im2col<<<dim3(HW / 64, 9), 256, 0, stream>>>(fc, offb, colT, 512, 4608);
            stsn_wmma_gemm<<<dim3(HW / 64, 512 / 128), 256, 0, stream>>>(A123[L], colT, fn, 4608);
            stsn_gn_relu<<<32, 256, 0, stream>>>(fn, gn_g[L], gn_b[L], 16);
            float* t = fc; fc = fn; fn = t;
        }
        // layer 4: offsets from fc (512ch); deform-conv applied to agg_supp (256ch)
        stsn_offconv<<<(18 * HW + 255) / 256, 256, 0, stream>>>(fc, off_w[3], off_b[3], offb, 512);
        stsn_im2col<<<dim3(HW / 64, 9), 256, 0, stream>>>(agg_supp, offb, colT, 256, 2304);
        stsn_wmma_gemm<<<dim3(HW / 64, 256 / 128), 256, 0, stream>>>(A4, colT, out_feat, 2304);
        stsn_gn_relu<<<32, 256, 0, stream>>>(out_feat, gn_g[3], gn_b[3], 8);
    };

    run_agg(refr, refr, ttb);   // tt = agg(reference, reference)
    run_agg(supp, refr, tkb);   // tk = agg(support, reference)

    stsn_blend<<<(HW + 255) / 256, 256, 0, stream>>>(ttb, tkb, (float*)d_out);
}
